// VertexTransformer_26010321945302
// MI455X (gfx1250) — compile-verified, matrix operand order checked by reading, not run on hardware
//
#include <hip/hip_runtime.h>
#include <hip/hip_bf16.h>
#include <math.h>
#include <stdint.h>

// ---------------------------------------------------------------------------
// Sizes (match reference)
// ---------------------------------------------------------------------------
constexpr int CB  = 4;      // batch
constexpr int CJ  = 6890;   // joints
constexpr int CD  = 256;    // hidden
constexpr int CN  = 2048;   // tokens
constexpr int CL  = 4;      // layers
constexpr int CH  = 8;      // heads
constexpr int CDH = 64;     // head dim
constexpr int CIN = 512;    // H*DH
constexpr int CFF = 1024;   // ffn hidden
constexpr int CM  = 13780;  // upsampled joints

typedef __bf16 bf16;
typedef __attribute__((ext_vector_type(16))) __bf16 v16bf;
typedef __attribute__((ext_vector_type(8)))  __bf16 v8bf;
typedef __attribute__((ext_vector_type(8)))  float  v8f;

typedef unsigned int u32x4 __attribute__((ext_vector_type(4)));
typedef int          i32x8 __attribute__((ext_vector_type(8)));
typedef int          i32x4 __attribute__((ext_vector_type(4)));

// ---------------------------------------------------------------------------
// Helpers
// ---------------------------------------------------------------------------
__device__ __forceinline__ float gelu_erf(float x) {
    return 0.5f * x * (1.0f + erff(x * 0.70710678118654752f));
}

// low 32 bits of a generic pointer to __shared__ == LDS byte offset (aperture
// mapping)
__device__ __forceinline__ unsigned lds_off(const void* p) {
    return (unsigned)(unsigned long long)(uintptr_t)p;
}

// Issue a TDM 2D tile load: tile_k contiguous bf16 elements per row,
// tile_rows rows with stride row_stride (elements).  Remaining extents give
// hardware OOB-zeroing at the right/bottom edges.  D# per CDNA5 ISA.
__device__ __forceinline__ void tdm_load_2d(unsigned lds_byte_off, const bf16* gsrc,
                                            int tile_k, int tile_rows,
                                            long long remain_k, long long remain_rows,
                                            long long row_stride)
{
    const unsigned long long ga = (unsigned long long)(uintptr_t)gsrc;
    u32x4 g0;
    g0[0] = 1u;                                       // count=1, user mode
    g0[1] = lds_byte_off;                             // lds_addr
    g0[2] = (unsigned)(ga & 0xFFFFFFFFull);           // global_addr[31:0]
    g0[3] = (unsigned)((ga >> 32) & 0x1FFFFFFull)     // global_addr[56:32]
          | (2u << 30);                               // type=2 ("image")
    // group1 bit layout (256b):
    //  [17:16] data_size=1 (2B)   [79:48] tensor_dim0   [111:80] tensor_dim1
    //  [127:112] tile_dim0        [143:128] tile_dim1   [207:160] dim0_stride
    const unsigned long long td0 = (unsigned long long)(remain_k  < 0 ? 0 : remain_k);
    const unsigned long long td1 = (unsigned long long)(remain_rows < 0 ? 0 : remain_rows);
    unsigned long long qw0 = (1ull << 16) | ((td0 & 0xFFFFull) << 48);
    unsigned long long qw1 = ((td0 >> 16) & 0xFFFFull)
                           | ((td1 & 0xFFFFFFFFull) << 16)
                           | ((unsigned long long)(tile_k & 0xFFFF) << 48);
    unsigned long long qw2 = (unsigned long long)(tile_rows & 0xFFFF)
                           | (((unsigned long long)row_stride & 0xFFFFFFFFull) << 32);
    unsigned long long qw3 = ((unsigned long long)row_stride >> 32) & 0xFFFFull;
    i32x8 g1;
    g1[0] = (int)(unsigned)(qw0 & 0xFFFFFFFFull);  g1[1] = (int)(unsigned)(qw0 >> 32);
    g1[2] = (int)(unsigned)(qw1 & 0xFFFFFFFFull);  g1[3] = (int)(unsigned)(qw1 >> 32);
    g1[4] = (int)(unsigned)(qw2 & 0xFFFFFFFFull);  g1[5] = (int)(unsigned)(qw2 >> 32);
    g1[6] = (int)(unsigned)(qw3 & 0xFFFFFFFFull);  g1[7] = (int)(unsigned)(qw3 >> 32);
    const i32x4 gz4 = {0, 0, 0, 0};                // groups 2/3 unused (2D)
    const i32x8 gz8 = {0, 0, 0, 0, 0, 0, 0, 0};
    __builtin_amdgcn_tensor_load_to_lds(g0, g1, gz4, gz4, gz8, 0);
}

// Load a 16x32 bf16 WMMA A/B fragment from LDS (CDNA5 ISA 16-bit layout):
// lanes 0-15: rows 0-15, K 0..7 + 16..23; lanes 16-31: K 8..15 + 24..31.
__device__ __forceinline__ v16bf frag_ab(const bf16* base, int stride) {
    int lane = threadIdx.x & 31;
    int r    = lane & 15;
    int kb   = (lane >> 4) * 8;
    const bf16* p = base + r * stride + kb;
    v8bf lo = *(const v8bf*)(p);
    v8bf hi = *(const v8bf*)(p + 16);
    v16bf a;
#pragma unroll
    for (int i = 0; i < 8; ++i) { a[i] = lo[i]; a[i + 8] = hi[i]; }
    return a;
}

// ---------------------------------------------------------------------------
// Tiled GEMM, double-buffered TDM pipeline:
//   C[M,Ncol] = A[M,K] (bf16, row major) * Bt[Ncol,K]^T (bf16, [N,K] row major)
// Block tile 128x128, K tile 32, 8 waves.  Wave 0 issues the NEXT K-tile's
// two TDM loads into the alternate LDS buffers, then waits tensorcnt<=2:
// tensor ops retire in order, so the previous tile's 2 loads are complete
// while the new pair stays in flight behind the WMMA work.
// ---------------------------------------------------------------------------
#define EP_BIAS_COL 1   // += biasCol[n]
#define EP_BIAS_ROW 2   // += biasRow[m]
#define EP_EXTRA    4   // += extra[m*ldc + n]   (residual or pos-emb, fp32)
#define EP_GELU     8   // gelu(.)  (applied last)

template <int EPI, bool OBF>
__global__ __launch_bounds__(256) void gemm_wmma_tdm(
    const bf16* __restrict__ A,  long long sAb, long long lda,
    const bf16* __restrict__ Bt, long long sBb, long long ldb,
    float* __restrict__ Cf, bf16* __restrict__ Cb, long long sCb, int ldc,
    const float* __restrict__ biasCol,
    const float* __restrict__ biasRow,
    const float* __restrict__ extra, long long sEb,
    int M, int Ncol, int K)
{
    __shared__ __align__(16) bf16 As[2][128 * 32];   // [m][k], pitch 32 (TDM dest)
    __shared__ __align__(16) bf16 Bs[2][128 * 32];   // [n][k], pitch 32 (TDM dest)

    const int z = blockIdx.z;
    A  += (long long)z * sAb;
    Bt += (long long)z * sBb;
    if (Cf) Cf += (long long)z * sCb;
    if (Cb) Cb += (long long)z * sCb;
    const float* Ex = extra ? (extra + (long long)z * sEb) : nullptr;

    const int m0 = blockIdx.x * 128;
    const int n0 = blockIdx.y * 128;
    const int t    = threadIdx.x;
    const int w    = t >> 5;
    const int lane = t & 31;
    const int wm = (w >> 2) * 64;   // wave M offset (2 waves in M)
    const int wn = (w & 3) * 32;    // wave N offset (4 waves in N)

    v8f acc[4][2];
    const v8f zero8 = {0.f, 0.f, 0.f, 0.f, 0.f, 0.f, 0.f, 0.f};
#pragma unroll
    for (int mt = 0; mt < 4; ++mt)
#pragma unroll
        for (int nt = 0; nt < 2; ++nt) acc[mt][nt] = zero8;

    const unsigned ldsA0 = lds_off(&As[0][0]), ldsA1 = lds_off(&As[1][0]);
    const unsigned ldsB0 = lds_off(&Bs[0][0]), ldsB1 = lds_off(&Bs[1][0]);

    const int ktiles = (K + 31) >> 5;
    if (w == 0) {   // prologue: tile 0 -> buffer 0
        tdm_load_2d(ldsA0, A + (long long)m0 * lda, 32, 128, K, M - m0, lda);
        tdm_load_2d(ldsB0, Bt + (long long)n0 * ldb, 32, 128, K, Ncol - n0, ldb);
    }

    for (int kt = 0; kt < ktiles; ++kt) {
        const int cur = kt & 1;
        if (w == 0) {
            if (kt + 1 < ktiles) {   // issue next tile into alternate buffers
                const int kn = (kt + 1) * 32;
                tdm_load_2d(cur ? ldsA0 : ldsA1, A + (long long)m0 * lda + kn,
                            32, 128, K - kn, M - m0, lda);
                tdm_load_2d(cur ? ldsB0 : ldsB1, Bt + (long long)n0 * ldb + kn,
                            32, 128, K - kn, Ncol - n0, ldb);
                __builtin_amdgcn_s_wait_tensorcnt(2);  // tile kt landed (in-order)
            } else {
                __builtin_amdgcn_s_wait_tensorcnt(0);
            }
        }
        __syncthreads();   // publish buffer `cur`

        const bf16* Ab = &As[cur][0];
        const bf16* Bb = &Bs[cur][0];
        v16bf afr[4], bfr[2];
#pragma unroll
        for (int mt = 0; mt < 4; ++mt) afr[mt] = frag_ab(Ab + (wm + mt * 16) * 32, 32);
#pragma unroll
        for (int nt = 0; nt < 2; ++nt) bfr[nt] = frag_ab(Bb + (wn + nt * 16) * 32, 32);
#pragma unroll
        for (int mt = 0; mt < 4; ++mt)
#pragma unroll
            for (int nt = 0; nt < 2; ++nt)
                acc[mt][nt] = __builtin_amdgcn_wmma_f32_16x16x32_bf16(
                    false, afr[mt], false, bfr[nt], (short)0, acc[mt][nt],
                    false, false);
        __syncthreads();   // all waves done with `cur` before it is re-targeted
    }

    // epilogue + store (C layout: element r -> row (lane/16)*8+r, col lane&15)
    const int rbase = (lane >> 4) * 8;
    const int cl    = lane & 15;
#pragma unroll
    for (int mt = 0; mt < 4; ++mt) {
#pragma unroll
        for (int nt = 0; nt < 2; ++nt) {
#pragma unroll
            for (int r = 0; r < 8; ++r) {
                const int gm = m0 + wm + mt * 16 + rbase + r;
                const int gn = n0 + wn + nt * 16 + cl;
                if (gm < M && gn < Ncol) {
                    float v = acc[mt][nt][r];
                    if (EPI & EP_BIAS_COL) v += biasCol[gn];
                    if (EPI & EP_BIAS_ROW) v += biasRow[gm];
                    if (EPI & EP_EXTRA)    v += Ex[(long long)gm * ldc + gn];
                    if (EPI & EP_GELU)     v = gelu_erf(v);
                    if (OBF) Cb[(long long)gm * ldc + gn] = (bf16)v;
                    else     Cf[(long long)gm * ldc + gn] = v;
                }
            }
        }
    }
}

// ---------------------------------------------------------------------------
// Flash attention (bf16 in / bf16 out).  One block = 64 query rows of one
// (b,h); 4 waves, each owns a 16-row query stripe.  K tiles stream in via TDM
// (overlapped with the manual V-transpose loads), QK^T and PV via bf16 WMMA,
// online softmax via 16-lane shfl reductions.
// ---------------------------------------------------------------------------
__global__ __launch_bounds__(128) void flash_attn_k(
    const bf16* __restrict__ qkv,   // [B, N, 3*INNER]
    bf16* __restrict__ ob)          // [B, N, INNER]
{
    __shared__ __align__(16) bf16 Qs[64][72];     // [q][dh]
    __shared__ __align__(16) bf16 Ks[64 * 64];    // [key][dh], TDM dest, pitch 64
    __shared__ __align__(16) bf16 Vt[64][72];     // [dh][key]
    __shared__ __align__(16) bf16 Ps[4][16][72];  // per-wave P staging [q][key]

    const int qb = blockIdx.x * 64;
    const int h  = blockIdx.y;
    const int b  = blockIdx.z;
    const int t    = threadIdx.x;
    const int w    = t >> 5;
    const int lane = t & 31;

    const long long rowstride = 3LL * CIN;
    const long long base = (long long)b * CN * rowstride;
    const int qofs = h * CDH;
    const int kofs = CIN + h * CDH;
    const int vofs = 2 * CIN + h * CDH;

    // load Q tile once (b128 vector loads)
    {
        const int qr = t >> 1, c0 = (t & 1) * 32;
        const bf16* src = qkv + base + (long long)(qb + qr) * rowstride + qofs + c0;
#pragma unroll
        for (int e = 0; e < 4; ++e)
            *(v8bf*)&Qs[qr][c0 + e * 8] = *(const v8bf*)(src + e * 8);
    }
    __syncthreads();

    v16bf qf[2];
    qf[0] = frag_ab(&Qs[w * 16][0], 72);
    qf[1] = frag_ab(&Qs[w * 16][0] + 32, 72);

    const v8f zero8 = {0.f, 0.f, 0.f, 0.f, 0.f, 0.f, 0.f, 0.f};
    v8f o_[4];
#pragma unroll
    for (int nt = 0; nt < 4; ++nt) o_[nt] = zero8;
    float mrow[8], lrow[8];
#pragma unroll
    for (int r = 0; r < 8; ++r) { mrow[r] = -3.0e38f; lrow[r] = 0.f; }

    const int prow = (lane >> 4) * 8;
    const int pc   = lane & 15;
    const unsigned ldsK = lds_off(&Ks[0]);

    for (int ktile = 0; ktile < CN; ktile += 64) {
        // K tile [key][dh] via Tensor Data Mover (overlaps V manual loads)
        if (w == 0) tdm_load_2d(ldsK, qkv + base + (long long)ktile * rowstride + kofs,
                                CDH, 64, CDH, CN - ktile, rowstride);
        // V tile transposed [dh][key] (manual: needs transpose)
        {
            const int kr = t >> 1, c0 = (t & 1) * 32;
            const bf16* vsrc = qkv + base + (long long)(ktile + kr) * rowstride + vofs + c0;
            v8bf vv[4];
#pragma unroll
            for (int e = 0; e < 4; ++e) vv[e] = *(const v8bf*)(vsrc + e * 8);
#pragma unroll
            for (int e = 0; e < 4; ++e)
#pragma unroll
                for (int q = 0; q < 8; ++q) Vt[c0 + e * 8 + q][kr] = vv[e][q];
        }
        if (w == 0) __builtin_amdgcn_s_wait_tensorcnt(0);
        __syncthreads();

        // S = (Q K^T) * scale   -- 16 queries x 64 keys per wave
        v8f s[4];
#pragma unroll
        for (int nt = 0; nt < 4; ++nt) s[nt] = zero8;
#pragma unroll
        for (int kc = 0; kc < 2; ++kc) {
#pragma unroll
            for (int nt = 0; nt < 4; ++nt) {
                v16bf kf = frag_ab(&Ks[(nt * 16) * 64] + kc * 32, 64);
                s[nt] = __builtin_amdgcn_wmma_f32_16x16x32_bf16(
                    false, qf[kc], false, kf, (short)0, s[nt], false, false);
            }
        }
#pragma unroll
        for (int nt = 0; nt < 4; ++nt) s[nt] = s[nt] * 0.125f;   // DH^-0.5

        // online softmax: row max / rescale
#pragma unroll
        for (int r = 0; r < 8; ++r) {
            float v = fmaxf(fmaxf(s[0][r], s[1][r]), fmaxf(s[2][r], s[3][r]));
            v = fmaxf(v, __shfl_xor(v, 1, 32));
            v = fmaxf(v, __shfl_xor(v, 2, 32));
            v = fmaxf(v, __shfl_xor(v, 4, 32));
            v = fmaxf(v, __shfl_xor(v, 8, 32));
            const float mnew  = fmaxf(mrow[r], v);
            const float alpha = __expf(mrow[r] - mnew);
            lrow[r] *= alpha;
#pragma unroll
            for (int nt = 0; nt < 4; ++nt) o_[nt][r] = o_[nt][r] * alpha;
            mrow[r] = mnew;
        }

        // P = exp(S - m) -> LDS (A-fragment order); accumulate row sums
#pragma unroll
        for (int r = 0; r < 8; ++r) {
            float psum = 0.f;
#pragma unroll
            for (int nt = 0; nt < 4; ++nt) {
                const float p = __expf(s[nt][r] - mrow[r]);
                psum += p;
                Ps[w][prow + r][nt * 16 + pc] = (bf16)p;
            }
            psum += __shfl_xor(psum, 1, 32);
            psum += __shfl_xor(psum, 2, 32);
            psum += __shfl_xor(psum, 4, 32);
            psum += __shfl_xor(psum, 8, 32);
            lrow[r] += psum;
        }

        // O += P V
#pragma unroll
        for (int kc = 0; kc < 2; ++kc) {
            v16bf pf = frag_ab(&Ps[w][0][0] + kc * 32, 72);
#pragma unroll
            for (int nt = 0; nt < 4; ++nt) {
                v16bf vf = frag_ab(&Vt[nt * 16][0] + kc * 32, 72);
                o_[nt] = __builtin_amdgcn_wmma_f32_16x16x32_bf16(
                    false, pf, false, vf, (short)0, o_[nt], false, false);
            }
        }
        __syncthreads();   // protect Ks/Vt before next tile load
    }

    // write O / l  ->  [B,N,INNER] (bf16)
#pragma unroll
    for (int nt = 0; nt < 4; ++nt) {
#pragma unroll
        for (int r = 0; r < 8; ++r) {
            const int n    = qb + w * 16 + prow + r;
            const int dcol = nt * 16 + pc;
            ob[((long long)b * CN + n) * CIN + h * CDH + dcol] =
                (bf16)(o_[nt][r] / lrow[r]);
        }
    }
}

// ---------------------------------------------------------------------------
// Small kernels
// ---------------------------------------------------------------------------
// fp32 -> bf16 plain convert
__global__ void cvt_k(const float* __restrict__ s, bf16* __restrict__ d, long long n)
{
    long long i = ((long long)blockIdx.x * 256 + threadIdx.x) * 4;
    if (i + 3 < n) {
        const float4 v = *(const float4*)(s + i);
        d[i + 0] = (bf16)v.x; d[i + 1] = (bf16)v.y;
        d[i + 2] = (bf16)v.z; d[i + 3] = (bf16)v.w;
    } else {
        for (long long e = i; e < n; ++e) d[e] = (bf16)s[e];
    }
}

// fp32 [R,C] -> bf16 transposed [C,R]; blockIdx.z batches planes
__global__ void cvt_tr_k(const float* __restrict__ s, bf16* __restrict__ d, int R, int C)
{
    const long long plane = (long long)R * C;
    s += blockIdx.z * plane;
    d += blockIdx.z * plane;
    const long long i = (long long)blockIdx.x * 256 + threadIdx.x;
    if (i >= plane) return;
    const int r = (int)(i / C), c = (int)(i % C);
    d[(long long)c * R + r] = (bf16)s[i];
}

// token embed, written TRANSPOSED as bf16: embT[b][d][j]  (B operand of pre_conv)
__global__ void pre_embed_k(const float* __restrict__ x, const float* __restrict__ wgt,
                            const float* __restrict__ bias, bf16* __restrict__ embT,
                            int total)
{
    int i = blockIdx.x * 256 + threadIdx.x;
    if (i >= total) return;
    const int d  = i & (CD - 1);
    const int bj = i >> 8;
    const int j  = bj % CJ;
    const int b  = bj / CJ;
    const float* xp = x + (long long)bj * 3;
    const float v = xp[0] * wgt[d] + xp[1] * wgt[CD + d] + xp[2] * wgt[2 * CD + d] + bias[d];
    embT[((long long)b * CD + d) * CJ + j] = (bf16)v;
}

// layernorm over last dim (D=256); writes fp32 and/or bf16 (non-null outputs)
__global__ __launch_bounds__(256) void layernorm_k(
    const float* __restrict__ in, float* __restrict__ outf, bf16* __restrict__ outb,
    const float* __restrict__ g, const float* __restrict__ be)
{
    __shared__ float sh[256];
    const long long row = blockIdx.x;
    const int tid = threadIdx.x;
    const float v = in[row * CD + tid];

    sh[tid] = v;
    __syncthreads();
    for (int o = 128; o > 0; o >>= 1) {
        if (tid < o) sh[tid] += sh[tid + o];
        __syncthreads();
    }
    const float mu = sh[0] * (1.0f / CD);
    __syncthreads();
    const float dvi = v - mu;
    sh[tid] = dvi * dvi;
    __syncthreads();
    for (int o = 128; o > 0; o >>= 1) {
        if (tid < o) sh[tid] += sh[tid + o];
        __syncthreads();
    }
    const float var = sh[0] * (1.0f / CD);
    const float r = dvi * rsqrtf(var + 1e-5f) * g[tid] + be[tid];
    if (outf) outf[row * CD + tid] = r;
    if (outb) outb[row * CD + tid] = (bf16)r;
}

// head output: out[row, colofs+o] = hid[row,:] . w2[:,o] + b2[o]
__global__ void head_out_k(const float* __restrict__ hid, const float* __restrict__ w2,
                           const float* __restrict__ b2, float* __restrict__ out,
                           int width, int colofs, int rows)
{
    const int i = blockIdx.x * 256 + threadIdx.x;
    if (i >= rows * width) return;
    const int row = i / width;
    const int o   = i - row * width;
    const float* hp = hid + (long long)row * CD;
    float s = b2[o];
#pragma unroll 8
    for (int e = 0; e < CD; ++e) s = fmaf(hp[e], w2[e * 4 + o], s);
    out[(long long)row * 14 + colofs + o] = s;
}

// ---------------------------------------------------------------------------
// Orchestration
// ---------------------------------------------------------------------------
extern "C" void kernel_launch(void* const* d_in, const int* in_sizes, int n_in,
                              void* d_out, int out_size, void* d_ws, size_t ws_size,
                              hipStream_t stream)
{
    (void)in_sizes; (void)n_in; (void)out_size; (void)ws_size;

    const float* x          = (const float*)d_in[0];
    const float* pre_emb_w  = (const float*)d_in[1];
    const float* pre_emb_b  = (const float*)d_in[2];
    const float* pre_conv_w = (const float*)d_in[3];
    const float* pre_conv_b = (const float*)d_in[4];
    const float* pos_emb    = (const float*)d_in[5];
    const float* pre_norm_g = (const float*)d_in[6];
    const float* pre_norm_b = (const float*)d_in[7];
    const float* ln1_g      = (const float*)d_in[8];
    const float* ln1_b      = (const float*)d_in[9];
    const float* qkv_w      = (const float*)d_in[10];
    const float* out_w      = (const float*)d_in[11];
    const float* out_b      = (const float*)d_in[12];
    const float* ln2_g      = (const float*)d_in[13];
    const float* ln2_b      = (const float*)d_in[14];
    const float* ff_w1      = (const float*)d_in[15];
    const float* ff_b1      = (const float*)d_in[16];
    const float* ff_w2      = (const float*)d_in[17];
    const float* ff_b2      = (const float*)d_in[18];
    const float* up_w       = (const float*)d_in[19];
    const float* up_b       = (const float*)d_in[20];
    const float* heads_w1   = (const float*)d_in[21];
    const float* heads_b1   = (const float*)d_in[22];
    const float* heads_w2   = (const float*)d_in[23];
    const float* heads_b2   = (const float*)d_in[24];
    float* out = (float*)d_out;

    // ---- workspace carve-out: fp32 region then bf16 region ----
    float* buf_h   = (float*)d_ws;                       // [B,N,D]   fp32
    float* buf_hid = buf_h + (size_t)CB * CN * CD;       // [B*M,D]   fp32
    bf16*  bp      = (bf16*)(buf_hid + (size_t)CB * CM * CD);
    bf16* wT_qkv = bp;  bp += (size_t)CL * 3 * CIN * CD;   // [L][3IN,D]
    bf16* wT_out = bp;  bp += (size_t)CL * CD * CIN;       // [L][D,IN]
    bf16* wT_ff1 = bp;  bp += (size_t)CL * CFF * CD;       // [L][FF,D]
    bf16* wT_ff2 = bp;  bp += (size_t)CL * CD * CFF;       // [L][D,FF]
    bf16* wT_h1  = bp;  bp += (size_t)5 * CD * CD;         // [5][D,D]
    bf16* w_pc   = bp;  bp += (size_t)CN * CJ;             // [N,J]
    bf16* w_up   = bp;  bp += (size_t)CM * CN;             // [M,N]
    bf16* embT   = bp;  bp += (size_t)CB * CD * CJ;        // [B][D,J]
    bf16* y_bf   = bp;  bp += (size_t)CB * CN * CD;        // [B,N,D]
    bf16* qkv_bf = bp;  bp += (size_t)CB * CN * 3 * CIN;   // [B,N,3IN]
    bf16* ao_bf  = bp;  bp += (size_t)CB * CN * CIN;       // [B,N,IN]
    bf16* ff_bf  = bp;  bp += (size_t)CB * CN * CFF;       // [B,N,FF]
    bf16* u_bf   = bp;  bp += (size_t)CB * CM * CD;        // [B,M,D]
    bf16* hT_bf  = bp;                                     // [B][D,N]

    // ---- one-time weight conversions (bf16; B-side transposed to [N,K]) ----
    {
        long long pl;
        pl = (long long)CD * 3 * CIN;
        cvt_tr_k<<<dim3((unsigned)((pl + 255) / 256), 1, CL), 256, 0, stream>>>(qkv_w, wT_qkv, CD, 3 * CIN);
        pl = (long long)CIN * CD;
        cvt_tr_k<<<dim3((unsigned)((pl + 255) / 256), 1, CL), 256, 0, stream>>>(out_w, wT_out, CIN, CD);
        pl = (long long)CD * CFF;
        cvt_tr_k<<<dim3((unsigned)((pl + 255) / 256), 1, CL), 256, 0, stream>>>(ff_w1, wT_ff1, CD, CFF);
        pl = (long long)CFF * CD;
        cvt_tr_k<<<dim3((unsigned)((pl + 255) / 256), 1, CL), 256, 0, stream>>>(ff_w2, wT_ff2, CFF, CD);
        pl = (long long)CD * CD;
        cvt_tr_k<<<dim3((unsigned)((pl + 255) / 256), 1, 5), 256, 0, stream>>>(heads_w1, wT_h1, CD, CD);
        pl = (long long)CN * CJ;
        cvt_k<<<(unsigned)((pl + 1023) / 1024), 256, 0, stream>>>(pre_conv_w, w_pc, pl);
        pl = (long long)CM * CN;
        cvt_k<<<(unsigned)((pl + 1023) / 1024), 256, 0, stream>>>(up_w, w_up, pl);
    }

    // 1) token embedding -> embT (bf16, transposed per batch)
    {
        const int total = CB * CJ * CD;
        pre_embed_k<<<(total + 255) / 256, 256, 0, stream>>>(x, pre_emb_w, pre_emb_b, embT, total);
    }

    // 2) pre-conv GEMM -> buf_h (fp32) + row bias + pos-emb
    {
        dim3 g((CN + 127) / 128, (CD + 127) / 128, CB);
        gemm_wmma_tdm<EP_BIAS_ROW | EP_EXTRA, false><<<g, 256, 0, stream>>>(
            w_pc, 0, CJ,
            embT, (long long)CD * CJ, CJ,
            buf_h, nullptr, (long long)CN * CD, CD,
            nullptr, pre_conv_b, pos_emb, 0,
            CN, CD, CJ);
    }

    // 3) pre layernorm (fp32 in place)
    layernorm_k<<<CB * CN, 256, 0, stream>>>(buf_h, buf_h, nullptr, pre_norm_g, pre_norm_b);

    // 4) transformer layers
    for (int i = 0; i < CL; ++i) {
        layernorm_k<<<CB * CN, 256, 0, stream>>>(buf_h, nullptr, y_bf,
                                                 ln1_g + i * CD, ln1_b + i * CD);
        {   // QKV -> bf16
            dim3 g((CN + 127) / 128, (3 * CIN + 127) / 128, CB);
            gemm_wmma_tdm<0, true><<<g, 256, 0, stream>>>(
                y_bf, (long long)CN * CD, CD,
                wT_qkv + (size_t)i * 3 * CIN * CD, 0, CD,
                nullptr, qkv_bf, (long long)CN * 3 * CIN, 3 * CIN,
                nullptr, nullptr, nullptr, 0,
                CN, 3 * CIN, CD);
        }
        flash_attn_k<<<dim3(CN / 64, CH, CB), 128, 0, stream>>>(qkv_bf, ao_bf);
        {   // out projection + bias + residual -> buf_h (fp32)
            dim3 g((CN + 127) / 128, (CD + 127) / 128, CB);
            gemm_wmma_tdm<EP_BIAS_COL | EP_EXTRA, false><<<g, 256, 0, stream>>>(
                ao_bf, (long long)CN * CIN, CIN,
                wT_out + (size_t)i * CD * CIN, 0, CIN,
                buf_h, nullptr, (long long)CN * CD, CD,
                out_b + i * CD, nullptr, buf_h, (long long)CN * CD,
                CN, CD, CIN);
        }
        layernorm_k<<<CB * CN, 256, 0, stream>>>(buf_h, nullptr, y_bf,
                                                 ln2_g + i * CD, ln2_b + i * CD);
        {   // FFN up + GELU -> bf16
            dim3 g((CN + 127) / 128, (CFF + 127) / 128, CB);
            gemm_wmma_tdm<EP_BIAS_COL | EP_GELU, true><<<g, 256, 0, stream>>>(
                y_bf, (long long)CN * CD, CD,
                wT_ff1 + (size_t)i * CFF * CD, 0, CD,
                nullptr, ff_bf, (long long)CN * CFF, CFF,
                ff_b1 + i * CFF, nullptr, nullptr, 0,
                CN, CFF, CD);
        }
        {   // FFN down + bias + residual -> buf_h (fp32)
            dim3 g((CN + 127) / 128, (CD + 127) / 128, CB);
            gemm_wmma_tdm<EP_BIAS_COL | EP_EXTRA, false><<<g, 256, 0, stream>>>(
                ff_bf, (long long)CN * CFF, CFF,
                wT_ff2 + (size_t)i * CD * CFF, 0, CFF,
                buf_h, nullptr, (long long)CN * CD, CD,
                ff_b2 + i * CD, nullptr, buf_h, (long long)CN * CD,
                CN, CD, CFF);
        }
    }

    // 5) h^T (bf16) then upsample GEMM -> u_bf
    {
        long long pl = (long long)CN * CD;
        cvt_tr_k<<<dim3((unsigned)((pl + 255) / 256), 1, CB), 256, 0, stream>>>(buf_h, hT_bf, CN, CD);
        dim3 g((CM + 127) / 128, (CD + 127) / 128, CB);
        gemm_wmma_tdm<EP_BIAS_ROW, true><<<g, 256, 0, stream>>>(
            w_up, 0, CN,
            hT_bf, (long long)CD * CN, CN,
            nullptr, u_bf, (long long)CM * CD, CD,
            nullptr, up_b, nullptr, 0,
            CM, CD, CN);
    }

    // 6) five gaussian-attribute heads
    const int widths[5] = {3, 1, 3, 4, 3};
    const int ofs[5]    = {0, 3, 4, 7, 11};
    const int rows = CB * CM;
    for (int k = 0; k < 5; ++k) {
        dim3 g((rows + 127) / 128, (CD + 127) / 128, 1);
        gemm_wmma_tdm<EP_BIAS_COL | EP_GELU, false><<<g, 256, 0, stream>>>(
            u_bf, 0, CD,
            wT_h1 + (size_t)k * CD * CD, 0, CD,
            buf_hid, nullptr, 0, CD,
            heads_b1 + k * CD, nullptr, nullptr, 0,
            rows, CD, CD);
        const int tot = rows * widths[k];
        head_out_k<<<(tot + 255) / 256, 256, 0, stream>>>(
            buf_hid, heads_w2 + (size_t)k * CD * 4, heads_b2 + k * 4,
            out, widths[k], ofs[k], rows);
    }
}
